// GT_56126632624584
// MI455X (gfx1250) — matrix operand order repaired and checked by reading, compiled, not run on hardware
//
#include <hip/hip_runtime.h>

typedef __bf16 bf16;
typedef __attribute__((ext_vector_type(16))) __bf16 v16bf;
typedef __attribute__((ext_vector_type(8)))  __bf16 v8bf;
typedef __attribute__((ext_vector_type(8)))  float  v8f;

#define BB 4
#define NN 4096
#define KNN 16
#define PP 64
#define DD 128
#define M1 (BB*NN)      /* 16384  */
#define M2 (BB*NN*KNN)  /* 262144 */

__device__ __forceinline__ float bf2f(bf16 x){
  unsigned short s; __builtin_memcpy(&s, &x, 2);
  unsigned int u = ((unsigned int)s) << 16;
  float f; __builtin_memcpy(&f, &u, 4); return f;
}
__device__ __forceinline__ bf16 f2bf(float f){
  unsigned int u; __builtin_memcpy(&u, &f, 4);
  unsigned int r = (u + 0x7fffu + ((u >> 16) & 1u)) >> 16;
  unsigned short s = (unsigned short)r;
  bf16 b; __builtin_memcpy(&b, &s, 2); return b;
}

// ---------------- A-operand loaders (return fp32; converted to bf16 on LDS stage) -------
struct LdF32 {
  const float* A; int ld;
  static constexpr bool k_contig = true;
  __device__ float operator()(long r, int k) const { return A[r*(long)ld + k]; }
};
struct LdBf16A {
  const bf16* A; int ld;
  static constexpr bool k_contig = true;
  __device__ float operator()(long r, int k) const { return bf2f(A[r*(long)ld + k]); }
};
// knn_features [B,3P,N,K] -> row (b,n,kn), col k=channel.  N*K = 65536 per (b,c) plane.
struct LdKnn {
  const float* kn;
  static constexpr bool k_contig = false;  // rows are contiguous in memory, k is strided
  __device__ float operator()(long r, int k) const {
    long b = r >> 16; long rem = r & 65535;
    return kn[(b*192 + k)*65536 + rem];
  }
};
// gamma input h = q[b,n,:] - k[b,idx,:] + kf[row,:]
struct LdGamma {
  const float* q; const float* kk; const bf16* kf; const int* idx;
  static constexpr bool k_contig = true;
  __device__ float operator()(long r, int k) const {
    long bn = r >> 4; long b = r >> 16; int id = idx[r];
    return q[bn*DD + k] - kk[(b*(long)NN + id)*DD + k] + bf2f(kf[r*DD + k]);
  }
};
struct LdDiff {
  const float* a; const float* b2;
  static constexpr bool k_contig = true;
  __device__ float operator()(long r, int k) const { return a[r*DD + k] - b2[r*DD + k]; }
};

// ---------------- Generic bf16 WMMA GEMM:  C[M,NDIM] = act(A[M,KDIM] @ W[KDIM,NDIM] + bias)
// Block covers 16*MT rows; each wave owns a 16-wide N-tile and keeps all its B (weight)
// fragments register-resident across the MT accumulator tiles.
template <int KDIM, int NDIM, int MT, typename ALd>
__global__ __launch_bounds__(2*NDIM)
void gemm_k(ALd ald, const bf16* __restrict__ W, const float* __restrict__ bias,
            float* __restrict__ outF, bf16* __restrict__ outB, int relu_flag) {
  constexpr int NT = 2*NDIM;               // threads per block (NDIM/16 waves)
  constexpr int KT = KDIM/32;              // k-steps
  constexpr int MR = 16*MT;                // rows per block
  __shared__ bf16 As[MR*KDIM];
  const long m0 = (long)blockIdx.x * MR;
  const int tid = threadIdx.x;
  if (ALd::k_contig) {
    for (int i = tid; i < MR*KDIM; i += NT) {
      int r = i / KDIM, k = i - r*KDIM;    // consecutive tid -> consecutive k (row-major A)
      As[r*KDIM + k] = f2bf(ald(m0 + r, k));
    }
  } else {
    for (int i = tid; i < MR*KDIM; i += NT) {
      int k = i / MR, r = i - k*MR;        // consecutive tid -> consecutive rows (coalesced gather)
      As[r*KDIM + k] = f2bf(ald(m0 + r, k));
    }
  }
  __syncthreads();
  const int wave = tid >> 5, lane = tid & 31;
  const int half = lane >> 4, lr = lane & 15;
  const int n0 = wave * 16;
  const int col = n0 + lr;

  // Preload B fragments for all k-steps (register resident; reused by MT tiles).
  v16bf bfr[KT];
#pragma unroll
  for (int kt = 0; kt < KT; ++kt) {
    const bf16* wp = &W[(kt*32 + half*16)*NDIM + col];
#pragma unroll
    for (int i = 0; i < 16; ++i) bfr[kt][i] = wp[i*NDIM];
  }
  const float bv = bias ? bias[col] : 0.f;

#pragma unroll
  for (int mt = 0; mt < MT; ++mt) {
    v8f acc = {};
#pragma unroll
    for (int kt = 0; kt < KT; ++kt) {
      // A fragment: lane row (mt*16+lr), K = half*16 + i (contiguous)
      const bf16* ap = &As[(mt*16 + lr)*KDIM + kt*32 + half*16];
      v8bf alo = *(const v8bf*)(ap);
      v8bf ahi = *(const v8bf*)(ap + 8);
      v16bf a;
#pragma unroll
      for (int i = 0; i < 8; ++i) { a[i] = alo[i]; a[8+i] = ahi[i]; }
      acc = __builtin_amdgcn_wmma_f32_16x16x32_bf16(false, a, false, bfr[kt], (short)0, acc, false, false);
    }
#pragma unroll
    for (int j = 0; j < 8; ++j) {
      float v = acc[j] + bv;
      if (relu_flag) v = fmaxf(v, 0.f);
      long grow = m0 + mt*16 + half*8 + j;
      if (outF) outF[grow*NDIM + col] = v;
      if (outB) outB[grow*NDIM + col] = f2bf(v);
    }
  }
}

// ---------------- weight fp32 -> bf16 (optional transpose): dst[k*Ncol+n] -----------------
__global__ void conv_w_k(const float* __restrict__ src, bf16* __restrict__ dst,
                         int K, int Ncol, int trans) {
  int i = blockIdx.x*256 + threadIdx.x;
  if (i >= K*Ncol) return;
  int k = i / Ncol, n = i - k*Ncol;
  float v = trans ? src[n*K + k] : src[i];
  dst[i] = f2bf(v);
}

// ---------------- log-softmax over K + weighted reduce -> res  (one block per (b,n)) ------
__global__ __launch_bounds__(128)
void attn_res_k(float* __restrict__ h, const bf16* __restrict__ kf,
                const float* __restrict__ v, const int* __restrict__ idx,
                float* __restrict__ res) {
  long bn = blockIdx.x; long b = bn >> 12;
  int d = threadIdx.x;
  const float inv = 0.0883883476483184f;   // 1/sqrt(128)
  __shared__ int sidx[KNN];
  if (d < KNN) sidx[d] = idx[bn*KNN + d];
  __syncthreads();
  long base = bn * KNN;
  float hv[KNN];
#pragma unroll
  for (int k = 0; k < KNN; ++k) hv[k] = h[(base+k)*DD + d] * inv;
  float m = hv[0];
#pragma unroll
  for (int k = 1; k < KNN; ++k) m = fmaxf(m, hv[k]);
  float z = 0.f;
#pragma unroll
  for (int k = 0; k < KNN; ++k) z += __expf(hv[k] - m);
  float lz = __logf(z);
  float acc = 0.f;
#pragma unroll
  for (int k = 0; k < KNN; ++k) {
    float a = hv[k] - m - lz;
    h[(base+k)*DD + d] = a;                                   // attn output (in place)
    float vv = v[(b*(long)NN + sidx[k])*DD + d] + bf2f(kf[(base+k)*DD + d]);
    acc += a * vv;
  }
  res[bn*DD + d] = acc;
}

// ---------------- BatchNorm stats: one block per channel ---------------------------------
__global__ __launch_bounds__(256)
void bn_stats_k(const float* __restrict__ y, float* __restrict__ mean, float* __restrict__ rstd) {
  int ch = blockIdx.x, tid = threadIdx.x;
  float s = 0.f, sq = 0.f;
  for (int row = tid; row < M1; row += 256) {
    float v = y[(long)row*DD + ch]; s += v; sq += v*v;
  }
  __shared__ float ss[256], qq[256];
  ss[tid] = s; qq[tid] = sq; __syncthreads();
  for (int off = 128; off; off >>= 1) {
    if (tid < off) { ss[tid] += ss[tid+off]; qq[tid] += qq[tid+off]; }
    __syncthreads();
  }
  if (!tid) {
    float mu = ss[0] * (1.f/M1);
    float var = qq[0] * (1.f/M1) - mu*mu;
    mean[ch] = mu; rstd[ch] = rsqrtf(var + 1e-5f);
  }
}

__global__ __launch_bounds__(256)
void bn1_apply_k(const float* __restrict__ y, const float* __restrict__ mean,
                 const float* __restrict__ rstd, const float* __restrict__ g,
                 const float* __restrict__ bb, const float* __restrict__ x,
                 float* __restrict__ r, bf16* __restrict__ r_bf) {
  long i = (long)blockIdx.x*256 + threadIdx.x;
  int d = (int)(i & (DD-1));
  float v = g[d]*(y[i]-mean[d])*rstd[d] + bb[d];
  v = fmaxf(v, 0.f) + x[i];
  r[i] = v; r_bf[i] = f2bf(v);
}

__global__ __launch_bounds__(256)
void final_k(const float* __restrict__ y2, const float* __restrict__ mean,
             const float* __restrict__ rstd, const float* __restrict__ g,
             const float* __restrict__ bb, const float* __restrict__ r,
             float* __restrict__ out) {
  long i = (long)blockIdx.x*256 + threadIdx.x;
  int d = (int)(i & (DD-1));
  float v = g[d]*(y2[i]-mean[d])*rstd[d] + bb[d];
  out[i] = r[i] + fmaxf(v, 0.f);
}

// xv [b,n,d] bf16 -> xvT [b,d,n] bf16
__global__ __launch_bounds__(256)
void xvT_k(const bf16* __restrict__ xv, bf16* __restrict__ xvT) {
  long i = (long)blockIdx.x*256 + threadIdx.x;
  long b = i >> 19; long rem = i & 524287; long n = rem >> 7; int d = (int)(rem & 127);
  xvT[(b*DD + d)*NN + n] = xv[i];
}

// ---------------- SA pass 1: per-row softmax stats of S = p p^T ---------------------------
__global__ __launch_bounds__(128)
void sa_rowstats_k(const bf16* __restrict__ p, float* __restrict__ rowmax,
                   float* __restrict__ rowinv) {
  long bi = blockIdx.x; long b = bi >> 12;
  int tid = threadIdx.x;
  __shared__ float pi[32];
  __shared__ float Sr[NN];
  __shared__ float red[128];
  if (tid < 32) pi[tid] = bf2f(p[bi*32 + tid]);
  __syncthreads();
  for (int j = tid; j < NN; j += 128) {
    const bf16* pj = &p[(b*(long)NN + j)*32];
    float dot = 0.f;
#pragma unroll
    for (int o = 0; o < 32; ++o) dot += pi[o] * bf2f(pj[o]);
    Sr[j] = dot;
  }
  __syncthreads();
  float m = -1e30f;
  for (int j = tid; j < NN; j += 128) m = fmaxf(m, Sr[j]);
  red[tid] = m; __syncthreads();
  for (int off = 64; off; off >>= 1) { if (tid < off) red[tid] = fmaxf(red[tid], red[tid+off]); __syncthreads(); }
  m = red[0]; __syncthreads();
  float z = 0.f;
  for (int j = tid; j < NN; j += 128) z += __expf(Sr[j] - m);
  red[tid] = z; __syncthreads();
  for (int off = 64; off; off >>= 1) { if (tid < off) red[tid] += red[tid+off]; __syncthreads(); }
  if (!tid) { rowmax[bi] = m; rowinv[bi] = 1.f / red[0]; }
}

// ---------------- SA pass 2 (flash): x_r = (xv @ att) / (1e-9 + colsum) -------------------
// grid = B*(N/16); 8 waves; wave w owns d-tile w; att never materialized in HBM.
__global__ __launch_bounds__(256)
void sa_xr_k(const bf16* __restrict__ p, const bf16* __restrict__ xvT,
             const float* __restrict__ rowmax, const float* __restrict__ rowinv,
             float* __restrict__ xr) {
  long blk = blockIdx.x; long b = blk >> 8; int n0 = (int)(blk & 255) * 16;
  int tid = threadIdx.x;
  int wave = tid >> 5, lane = tid & 31, half = lane >> 4, lr = lane & 15;
  int d0 = wave * 16;
  __shared__ bf16 att_s[8][16*32];       // per-wave [col(16)][m(32)] staging, B-frag order
  bf16* as = &att_s[wave][0];

  // B fragment of S: B[o][c] = p[n0+c, o]; lane column lr, o = half*16 + i (contiguous row)
  v16bf bp;
  {
    const bf16* pr = &p[(b*(long)NN + n0 + lr)*32 + half*16];
    v8bf t0 = *(const v8bf*)pr; v8bf t1 = *(const v8bf*)(pr + 8);
#pragma unroll
    for (int i = 0; i < 8; ++i) { bp[i] = t0[i]; bp[8+i] = t1[i]; }
  }
  v8f U = {};
  float csum = 0.f;
  const float* rm = &rowmax[b*(long)NN];
  const float* ri = &rowinv[b*(long)NN];

  for (int mc = 0; mc < NN; mc += 32) {
#pragma unroll
    for (int t = 0; t < 2; ++t) {
      int mrow = mc + t*16;
      // A fragment of S: row = mrow+lr of p, K = half*16 + i (contiguous)
      v16bf ap;
      {
        const bf16* pr = &p[(b*(long)NN + mrow + lr)*32 + half*16];
        v8bf lo = *(const v8bf*)pr; v8bf hi = *(const v8bf*)(pr + 8);
#pragma unroll
        for (int i = 0; i < 8; ++i) { ap[i] = lo[i]; ap[8+i] = hi[i]; }
      }
      v8f S = {};
      S = __builtin_amdgcn_wmma_f32_16x16x32_bf16(false, ap, false, bp, (short)0, S, false, false);
#pragma unroll
      for (int j = 0; j < 8; ++j) {
        int mloc = t*16 + half*8 + j;
        int mg = mc + mloc;
        float a = __expf(S[j] - rm[mg]) * ri[mg];   // softmax att[mg][n0+lr]
        csum += a;
        as[lr*32 + mloc] = f2bf(a);
      }
    }
    // A fragment of U: rows d, K = m chunk (xvT row-major over m)
    v16bf ax;
    {
      const bf16* xp = &xvT[(b*(long)DD + d0 + lr)*NN + mc + half*16];
      v8bf lo = *(const v8bf*)xp; v8bf hi = *(const v8bf*)(xp + 8);
#pragma unroll
      for (int i = 0; i < 8; ++i) { ax[i] = lo[i]; ax[8+i] = hi[i]; }
    }
    // B fragment of U from LDS: lane column lr, k = half*16 + i (contiguous)
    v16bf ba;
    {
      const bf16* ar = &as[lr*32 + half*16];
      v8bf lo = *(const v8bf*)ar; v8bf hi = *(const v8bf*)(ar + 8);
#pragma unroll
      for (int i = 0; i < 8; ++i) { ba[i] = lo[i]; ba[8+i] = hi[i]; }
    }
    U = __builtin_amdgcn_wmma_f32_16x16x32_bf16(false, ax, false, ba, (short)0, U, false, false);
  }
  float s = csum + __shfl_xor(csum, 16, 32);         // full column sum (rows split across halves)
  float rinv = 1.f / (1e-9f + s);
#pragma unroll
  for (int j = 0; j < 8; ++j) {
    int dd = d0 + half*8 + j;
    xr[(b*(long)NN + n0 + lr)*DD + dd] = U[j] * rinv;
  }
}

// =========================================================================================
extern "C" void kernel_launch(void* const* d_in, const int* in_sizes, int n_in,
                              void* d_out, int out_size, void* d_ws, size_t ws_size,
                              hipStream_t stream) {
  (void)in_sizes; (void)n_in; (void)out_size; (void)ws_size;
  const float* features = (const float*)d_in[0];
  const float* knnf     = (const float*)d_in[1];
  const int*   knn_idx  = (const int*)  d_in[2];
  const float* fc1_w = (const float*)d_in[3];  const float* fc1_b = (const float*)d_in[4];
  const float* fc2_w = (const float*)d_in[5];  const float* fc2_b = (const float*)d_in[6];
  const float* bn_g  = (const float*)d_in[7];  const float* bn_b  = (const float*)d_in[8];
  const float* fb_w1 = (const float*)d_in[9];  const float* fb_b1 = (const float*)d_in[10];
  const float* fb_w2 = (const float*)d_in[11]; const float* fb_b2 = (const float*)d_in[12];
  const float* g_w1  = (const float*)d_in[13]; const float* g_b1  = (const float*)d_in[14];
  const float* g_w2  = (const float*)d_in[15]; const float* g_b2  = (const float*)d_in[16];
  const float* wq = (const float*)d_in[17];
  const float* wk = (const float*)d_in[18];
  const float* wv = (const float*)d_in[19];
  const float* sa_qk_w = (const float*)d_in[20];
  const float* sa_v_w  = (const float*)d_in[21]; const float* sa_v_b  = (const float*)d_in[22];
  const float* sa_tr_w = (const float*)d_in[23]; const float* sa_tr_b = (const float*)d_in[24];
  const float* sa_bn_g = (const float*)d_in[25]; const float* sa_bn_b = (const float*)d_in[26];

  float* outp = (float*)d_out;                    // [M1*DD]
  float* attn = outp + (size_t)M1*DD;             // [M2*DD]; also pre-softmax gamma output

  char* base = (char*)d_ws; size_t off = 0;
  auto carve = [&](size_t bytes)->char* {
    char* ptr = base + off; off = (off + bytes + 255) & ~(size_t)255; return ptr;
  };
  float* x    = (float*)carve((size_t)M1*DD*4);
  float* q    = (float*)carve((size_t)M1*DD*4);
  float* kg   = (float*)carve((size_t)M1*DD*4);
  float* vg   = (float*)carve((size_t)M1*DD*4);
  float* res  = (float*)carve((size_t)M1*DD*4);
  float* y    = (float*)carve((size_t)M1*DD*4);
  float* r    = (float*)carve((size_t)M1*DD*4);
  float* xr   = (float*)carve((size_t)M1*DD*4);
  bf16*  x_bf = (bf16*) carve((size_t)M1*DD*2);   // reused as xv_bf later
  bf16*  r_bf = (bf16*) carve((size_t)M1*DD*2);
  bf16*  xvT  = (bf16*) carve((size_t)M1*DD*2);
  bf16*  p_bf = (bf16*) carve((size_t)M1*32*2);
  float* rowmax = (float*)carve((size_t)M1*4);
  float* rowinv = (float*)carve((size_t)M1*4);
  float* mean1 = (float*)carve(DD*4); float* rstd1 = (float*)carve(DD*4);
  float* mean2 = (float*)carve(DD*4); float* rstd2 = (float*)carve(DD*4);
  bf16* wfc1 = (bf16*)carve(64*128*2);
  bf16* wfc2 = (bf16*)carve(128*128*2);
  bf16* wfb1 = (bf16*)carve(192*128*2);
  bf16* wfb2 = (bf16*)carve(128*128*2);
  bf16* wg1  = (bf16*)carve(128*128*2);
  bf16* wg2  = (bf16*)carve(128*128*2);
  bf16* wwq  = (bf16*)carve(128*128*2);
  bf16* wwk  = (bf16*)carve(128*128*2);
  bf16* wwv  = (bf16*)carve(128*128*2);
  bf16* wqkT = (bf16*)carve(128*32*2);
  bf16* wvT  = (bf16*)carve(128*128*2);
  bf16* wtrT = (bf16*)carve(128*128*2);
  bf16* h1 = (bf16*)carve((size_t)M2*DD*2);       // fb mid, then gamma mid
  bf16* kf = (bf16*)carve((size_t)M2*DD*2);

  auto cw = [&](const float* s, bf16* d, int K, int Nc, int tr) {
    conv_w_k<<<(K*Nc + 255)/256, 256, 0, stream>>>(s, d, K, Nc, tr);
  };
  cw(fc1_w, wfc1, 64, 128, 0);   cw(fc2_w, wfc2, 128, 128, 0);
  cw(fb_w1, wfb1, 192, 128, 0);  cw(fb_w2, wfb2, 128, 128, 0);
  cw(g_w1, wg1, 128, 128, 0);    cw(g_w2, wg2, 128, 128, 0);
  cw(wq, wwq, 128, 128, 0);      cw(wk, wwk, 128, 128, 0);  cw(wv, wwv, 128, 128, 0);
  cw(sa_qk_w, wqkT, 128, 32, 1); cw(sa_v_w, wvT, 128, 128, 1);
  cw(sa_tr_w, wtrT, 128, 128, 1);

  // x = features @ fc1_w + b        (also bf16 copy)
  gemm_k<64,128,1,LdF32><<<M1/16, 256, 0, stream>>>(LdF32{features, PP}, wfc1, fc1_b, x, x_bf, 0);
  // q/k/v projections
  gemm_k<128,128,1,LdBf16A><<<M1/16, 256, 0, stream>>>(LdBf16A{x_bf, DD}, wwq, nullptr, q,  nullptr, 0);
  gemm_k<128,128,1,LdBf16A><<<M1/16, 256, 0, stream>>>(LdBf16A{x_bf, DD}, wwk, nullptr, kg, nullptr, 0);
  gemm_k<128,128,1,LdBf16A><<<M1/16, 256, 0, stream>>>(LdBf16A{x_bf, DD}, wwv, nullptr, vg, nullptr, 0);
  // fb MLP (transpose fused into A gather); MT=4 -> register-resident weight fragments
  gemm_k<192,128,4,LdKnn><<<M2/64, 256, 0, stream>>>(LdKnn{knnf}, wfb1, fb_b1, nullptr, h1, 1);
  gemm_k<128,128,4,LdBf16A><<<M2/64, 256, 0, stream>>>(LdBf16A{h1, DD}, wfb2, fb_b2, nullptr, kf, 0);
  // gamma MLP (q - k_gather + kf fused into A loader); pre-softmax h -> attn region of d_out
  gemm_k<128,128,4,LdGamma><<<M2/64, 256, 0, stream>>>(LdGamma{q, kg, kf, knn_idx}, wg1, g_b1, nullptr, h1, 1);
  gemm_k<128,128,4,LdBf16A><<<M2/64, 256, 0, stream>>>(LdBf16A{h1, DD}, wg2, g_b2, attn, nullptr, 0);
  // log-softmax over K (in place) + weighted reduce -> res
  attn_res_k<<<M1, 128, 0, stream>>>(attn, kf, vg, knn_idx, res);
  // y = res @ fc2_w + b; BN1; r = relu(bn(y)) + x
  gemm_k<128,128,1,LdF32><<<M1/16, 256, 0, stream>>>(LdF32{res, DD}, wfc2, fc2_b, y, nullptr, 0);
  bn_stats_k<<<DD, 256, 0, stream>>>(y, mean1, rstd1);
  bn1_apply_k<<<(M1*DD)/256, 256, 0, stream>>>(y, mean1, rstd1, bn_g, bn_b, x, r, r_bf);
  // SA: p = r @ qk_w^T  (shared q/k proj);  xv = r @ v_w^T + b  (bf16, reuse x_bf)
  gemm_k<128,32,1,LdBf16A><<<M1/16, 64, 0, stream>>>(LdBf16A{r_bf, DD}, wqkT, nullptr, nullptr, p_bf, 0);
  gemm_k<128,128,1,LdBf16A><<<M1/16, 256, 0, stream>>>(LdBf16A{r_bf, DD}, wvT, sa_v_b, nullptr, x_bf, 0);
  xvT_k<<<(M1*DD)/256, 256, 0, stream>>>(x_bf, xvT);
  // flash SA: row stats then x_r
  sa_rowstats_k<<<M1, 128, 0, stream>>>(p_bf, rowmax, rowinv);
  sa_xr_k<<<BB*(NN/16), 256, 0, stream>>>(p_bf, xvT, rowmax, rowinv, xr);
  // t = (r - x_r) @ tr_w^T + b; BN2; out = r + relu(bn2(t))
  gemm_k<128,128,1,LdDiff><<<M1/16, 256, 0, stream>>>(LdDiff{r, xr}, wtrT, sa_tr_b, y, nullptr, 0);
  bn_stats_k<<<DD, 256, 0, stream>>>(y, mean2, rstd2);
  final_k<<<(M1*DD)/256, 256, 0, stream>>>(y, mean2, rstd2, sa_bn_g, sa_bn_b, r, outp);
}
